// CausalAttention_48507360641472
// MI455X (gfx1250) — matrix operand order repaired and checked by reading, not compiled, not run on hardware
//
#include <hip/hip_runtime.h>
#include <hip/hip_bf16.h>

// ---------------------------------------------------------------------------
// Types for CDNA5 WMMA (wave32, 16x16x32 bf16 -> f32 accum)
// ---------------------------------------------------------------------------
typedef __bf16 bf16_t;
typedef __attribute__((ext_vector_type(8)))  __bf16 v8bf;
typedef __attribute__((ext_vector_type(16))) __bf16 v16bf;
typedef __attribute__((ext_vector_type(8)))  float  v8f;

union V16 {
    v16bf v;
    v8bf  h[2];
};

static __device__ __forceinline__ v8f wmma_bf16(v16bf a, v16bf b, v8f c) {
    // (neg_a, A, neg_b, B, c_mod, C, reuse_a, reuse_b)
    return __builtin_amdgcn_wmma_f32_16x16x32_bf16(false, a, false, b,
                                                   (short)0, c, false, false);
}

static __device__ __forceinline__ v8f v8f_zero() {
    v8f z;
#pragma unroll
    for (int i = 0; i < 8; ++i) z[i] = 0.0f;
    return z;
}

// ---------------------------------------------------------------------------
// CDNA5 async global->LDS copy (GLOBAL_LOAD_ASYNC_TO_LDS_B128, ASYNCcnt).
// Each lane copies 16 bytes from its global address to its LDS offset.
// The LDS offset is the low 32 bits of the flat shared pointer (AS3 offset).
// ---------------------------------------------------------------------------
static __device__ __forceinline__ void async_g2l_b128(void* lds, const void* g) {
    unsigned ldsoff = (unsigned)(uintptr_t)lds;
    asm volatile("global_load_async_to_lds_b128 %0, %1, off"
                 :: "v"(ldsoff), "v"(g) : "memory");
}
static __device__ __forceinline__ void wait_async() {
    asm volatile("s_wait_asynccnt 0" ::: "memory");
}

// ---------------------------------------------------------------------------
// Problem constants (match reference)
// ---------------------------------------------------------------------------
static constexpr int BATCH  = 2;
static constexpr int SEQ    = 2048;
static constexpr int DMODEL = 2048;
static constexpr int HEADS  = 16;
static constexpr int DHEAD  = 128;
static constexpr int NQKV   = 3 * HEADS * DHEAD;   // 6144
static constexpr int MROWS  = BATCH * SEQ;         // 4096

// ---------------------------------------------------------------------------
// fp32 -> bf16 conversion
// ---------------------------------------------------------------------------
__global__ void f32_to_bf16_kernel(const float* __restrict__ in,
                                   bf16_t* __restrict__ out, int n) {
    int i = blockIdx.x * blockDim.x + threadIdx.x;
    int stride = gridDim.x * blockDim.x;
    for (; i < n; i += stride) out[i] = (bf16_t)in[i];
}

// ---------------------------------------------------------------------------
// bf16 GEMM: C[MxN] = A[MxK] * B[KxN]; A,B row-major bf16.
// 256 threads = 8 waves. Block tile 128x128, K-step 32.
// Wave w computes rows [w*16, w*16+16) x all 128 cols (8 WMMA C tiles).
// A tile staged via async global->LDS; B tile transposed via DS stores.
// ---------------------------------------------------------------------------
template <bool OUT_BF16>
__global__ __launch_bounds__(256) void gemm_bf16_kernel(
    const bf16_t* __restrict__ A, const bf16_t* __restrict__ B,
    void* __restrict__ Cout, int M, int N, int K) {
    __shared__ bf16_t lA[128 * 32];    // [row][k]
    __shared__ bf16_t lBt[128 * 32];   // [n][k]  (transposed)

    const int tid  = threadIdx.x;
    const int lane = tid & 31;
    const int wave = tid >> 5;
    const int half = lane >> 4;   // 0 or 1 (lane group)
    const int lrow = lane & 15;

    const int rowBase = blockIdx.y * 128;
    const int colBase = blockIdx.x * 128;

    v8f acc[8];
#pragma unroll
    for (int c = 0; c < 8; ++c) acc[c] = v8f_zero();

    // A loader: thread t loads row t/2, k-half (t&1)*16 (16 bf16 = 32B)
    const int aRow  = tid >> 1;
    const int aColH = (tid & 1) * 16;
    // B loader: thread t loads global row k=(t&31), cols (t>>5)*16..+15,
    // scatters transposed into lBt.
    const int bRowK  = tid & 31;
    const int bColBk = (tid >> 5) * 16;

    for (int k0 = 0; k0 < K; k0 += 32) {
        {
            const bf16_t* src = A + (size_t)(rowBase + aRow) * K + k0 + aColH;
            bf16_t* dst = &lA[aRow * 32 + aColH];
            async_g2l_b128(dst, src);           // 16B
            async_g2l_b128(dst + 8, src + 8);   // 16B
            if (k0 + 32 < K)
                __builtin_prefetch(src + 32, 0, 0);   // global_prefetch_b8
        }
        {
            const bf16_t* src = B + (size_t)(k0 + bRowK) * N + colBase + bColBk;
#pragma unroll
            for (int j = 0; j < 16; ++j)
                lBt[(bColBk + j) * 32 + bRowK] = src[j];
            if (k0 + 32 < K)
                __builtin_prefetch(src + (size_t)32 * N, 0, 0);
        }
        wait_async();
        __syncthreads();

        // A fragment (16x32): lane holds row lrow; 16 contiguous k at 16*half
        // (matches B fragment k order).
        V16 afr;
        {
            const bf16_t* ap = &lA[(wave * 16 + lrow) * 32];
            afr.h[0] = *(const v8bf*)(ap + 16 * half);
            afr.h[1] = *(const v8bf*)(ap + 16 * half + 8);
        }
#pragma unroll
        for (int c = 0; c < 8; ++c) {
            // B fragment: lane holds col c*16+lrow; 16 contiguous k at 16*half.
            V16 bfr;
            const bf16_t* bp = &lBt[(c * 16 + lrow) * 32 + 16 * half];
            bfr.h[0] = *(const v8bf*)bp;
            bfr.h[1] = *(const v8bf*)(bp + 8);
            acc[c] = wmma_bf16(afr.v, bfr.v, acc[c]);
        }
        __syncthreads();
    }

    // Epilogue. C layout: VGPR r, lane l -> M = r + 8*half, N = lrow.
#pragma unroll
    for (int c = 0; c < 8; ++c) {
        const int col = colBase + c * 16 + lrow;
#pragma unroll
        for (int r = 0; r < 8; ++r) {
            const int row = rowBase + wave * 16 + r + 8 * half;
            if constexpr (OUT_BF16)
                ((bf16_t*)Cout)[(size_t)row * N + col] = (bf16_t)acc[c][r];
            else
                ((float*)Cout)[(size_t)row * N + col] = acc[c][r];
        }
    }
}

// ---------------------------------------------------------------------------
// RoPE + split QKV[B,L,H,3D] (bf16) -> Q,K (rotated), V in [B,H,L,D] bf16.
// One thread per (b,h,l,pair).
// ---------------------------------------------------------------------------
__global__ void rope_split_kernel(const bf16_t* __restrict__ qkv,
                                  bf16_t* __restrict__ Q,
                                  bf16_t* __restrict__ Kb,
                                  bf16_t* __restrict__ V) {
    const int idx   = blockIdx.x * blockDim.x + threadIdx.x;
    const int total = BATCH * HEADS * SEQ * (DHEAD / 2);
    if (idx >= total) return;
    const int i = idx % (DHEAD / 2);
    const int l = (idx / (DHEAD / 2)) % SEQ;
    const int h = (idx / ((DHEAD / 2) * SEQ)) % HEADS;
    const int b = idx / ((DHEAD / 2) * SEQ * HEADS);

    const bf16_t* src = qkv + (size_t)(b * SEQ + l) * NQKV + h * (3 * DHEAD);
    const float q1 = (float)src[2 * i],             q2 = (float)src[2 * i + 1];
    const float k1 = (float)src[DHEAD + 2 * i],     k2 = (float)src[DHEAD + 2 * i + 1];
    const float v1 = (float)src[2 * DHEAD + 2 * i], v2 = (float)src[2 * DHEAD + 2 * i + 1];

    // inv_freq = 10000^(-(2i)/128)
    const float freq = __expf(-((float)(2 * i) / (float)DHEAD) * 9.2103403719761836f);
    const float ang  = (float)l * freq;
    const float s = __sinf(ang), c = __cosf(ang);

    const size_t o = ((size_t)((b * HEADS + h) * SEQ + l)) * DHEAD + 2 * i;
    Q[o]      = (bf16_t)(q1 * c - q2 * s);
    Q[o + 1]  = (bf16_t)(q1 * s + q2 * c);
    Kb[o]     = (bf16_t)(k1 * c - k2 * s);
    Kb[o + 1] = (bf16_t)(k1 * s + k2 * c);
    V[o]      = (bf16_t)v1;
    V[o + 1]  = (bf16_t)v2;
}

// ---------------------------------------------------------------------------
// Flash attention (causal). Block = 128 threads = 4 waves. Each wave owns a
// 16-row q-tile; block covers 64 q rows. Keys processed in 32-wide tiles
// staged in LDS (K via async copy, V transposed via DS). Online softmax fp32.
// Output O[B, L, H*D] bf16.
// ---------------------------------------------------------------------------
__global__ __launch_bounds__(128) void attn_kernel(
    const bf16_t* __restrict__ Q, const bf16_t* __restrict__ K,
    const bf16_t* __restrict__ V, bf16_t* __restrict__ O) {
    __shared__ bf16_t lK[32 * 128];       // [key][d]
    __shared__ bf16_t lVt[128 * 32];      // [d][key] (transposed)
    __shared__ bf16_t lP[4][16 * 32];     // per-wave P staging [qrow][key]

    const int tid  = threadIdx.x;
    const int lane = tid & 31;
    const int wave = tid >> 5;
    const int half = lane >> 4;
    const int lrow = lane & 15;

    const int bh = blockIdx.y;           // b*HEADS + h
    const int b  = bh >> 4;
    const int h  = bh & 15;
    const int qt = blockIdx.x;           // q-tile of 64 rows
    const size_t base = (size_t)bh * SEQ * DHEAD;
    const int q0 = qt * 64 + wave * 16;  // wave's first q row

    // Load Q fragments (16x128 = 4 chunks of 16x32) once, kept in registers.
    V16 qf[4];
    {
        const bf16_t* qp = Q + base + (size_t)(q0 + lrow) * DHEAD;
#pragma unroll
        for (int c = 0; c < 4; ++c) {
            qf[c].h[0] = *(const v8bf*)(qp + c * 32 + 16 * half);
            qf[c].h[1] = *(const v8bf*)(qp + c * 32 + 16 * half + 8);
        }
    }

    v8f o_acc[8];
#pragma unroll
    for (int c = 0; c < 8; ++c) o_acc[c] = v8f_zero();
    float m[8], lsum[8];
#pragma unroll
    for (int r = 0; r < 8; ++r) { m[r] = -1e30f; lsum[r] = 0.0f; }

    const float scale = 0.08838834764831845f;   // 1/sqrt(128)
    const int nkt = (qt + 1) * 2;               // 32-key tiles, causal bound

    for (int kt = 0; kt < nkt; ++kt) {
        const int kbase = kt * 32;
        // Cooperative tile load: K via async global->LDS, V transposed via DS.
        {
            const int key   = tid >> 2;          // 0..31
            const int dbase = (tid & 3) * 32;    // 0,32,64,96
            const bf16_t* kp = K + base + (size_t)(kbase + key) * DHEAD + dbase;
            bf16_t* kd = &lK[key * 128 + dbase];
            async_g2l_b128(kd,      kp);         // 16B
            async_g2l_b128(kd + 8,  kp + 8);
            async_g2l_b128(kd + 16, kp + 16);
            async_g2l_b128(kd + 24, kp + 24);
            const bf16_t* vp = V + base + (size_t)(kbase + key) * DHEAD + dbase;
#pragma unroll
            for (int j = 0; j < 32; ++j)
                lVt[(dbase + j) * 32 + key] = vp[j];
        }
        wait_async();
        __syncthreads();

        // S = Q Kt : two 16x16 tiles (keys kbase..+15 and kbase+16..+31).
        v8f s0 = v8f_zero(), s1 = v8f_zero();
#pragma unroll
        for (int c = 0; c < 4; ++c) {
            V16 kf0, kf1;
            const bf16_t* kp0 = &lK[(lrow) * 128 + c * 32 + 16 * half];
            kf0.h[0] = *(const v8bf*)kp0;
            kf0.h[1] = *(const v8bf*)(kp0 + 8);
            const bf16_t* kp1 = &lK[(16 + lrow) * 128 + c * 32 + 16 * half];
            kf1.h[0] = *(const v8bf*)kp1;
            kf1.h[1] = *(const v8bf*)(kp1 + 8);
            s0 = wmma_bf16(qf[c].v, kf0.v, s0);
            s1 = wmma_bf16(qf[c].v, kf1.v, s1);
        }

        // Scale, causal mask, online softmax update.
        float p0[8], p1[8];
#pragma unroll
        for (int r = 0; r < 8; ++r) {
            const int qrow = q0 + r + 8 * half;
            const int c0 = kbase + lrow;
            const int c1 = kbase + 16 + lrow;
            float v0 = s0[r] * scale; if (c0 > qrow) v0 = -1e30f;
            float v1 = s1[r] * scale; if (c1 > qrow) v1 = -1e30f;
            float rm = fmaxf(v0, v1);
            rm = fmaxf(rm, __shfl_xor(rm, 1, 32));
            rm = fmaxf(rm, __shfl_xor(rm, 2, 32));
            rm = fmaxf(rm, __shfl_xor(rm, 4, 32));
            rm = fmaxf(rm, __shfl_xor(rm, 8, 32));
            const float mn    = fmaxf(m[r], rm);
            const float alpha = __expf(m[r] - mn);
            const float e0 = __expf(v0 - mn);
            const float e1 = __expf(v1 - mn);
            float rs = e0 + e1;
            rs += __shfl_xor(rs, 1, 32);
            rs += __shfl_xor(rs, 2, 32);
            rs += __shfl_xor(rs, 4, 32);
            rs += __shfl_xor(rs, 8, 32);
            lsum[r] = lsum[r] * alpha + rs;
            m[r] = mn;
            p0[r] = e0; p1[r] = e1;
#pragma unroll
            for (int c = 0; c < 8; ++c) o_acc[c][r] *= alpha;
        }

        // Restage P (C-layout f32 -> A-layout bf16) through wave-private LDS.
        bf16_t* pw = &lP[wave][0];
#pragma unroll
        for (int r = 0; r < 8; ++r) {
            pw[(r + 8 * half) * 32 + lrow]      = (bf16_t)p0[r];
            pw[(r + 8 * half) * 32 + 16 + lrow] = (bf16_t)p1[r];
        }
        asm volatile("s_wait_dscnt 0" ::: "memory");  // wave-local LDS ordering

        V16 pf;
        {
            const bf16_t* pp = &pw[lrow * 32];
            pf.h[0] = *(const v8bf*)(pp + 16 * half);
            pf.h[1] = *(const v8bf*)(pp + 16 * half + 8);
        }
        // O += P (16x32) @ V (32x128)
#pragma unroll
        for (int c = 0; c < 8; ++c) {
            V16 vf;
            const bf16_t* vp2 = &lVt[(c * 16 + lrow) * 32 + 16 * half];
            vf.h[0] = *(const v8bf*)vp2;
            vf.h[1] = *(const v8bf*)(vp2 + 8);
            o_acc[c] = wmma_bf16(pf.v, vf.v, o_acc[c]);
        }
        __syncthreads();
    }

    // Epilogue: O[b, qrow, h*128 + d] = o_acc / lsum
#pragma unroll
    for (int r = 0; r < 8; ++r) {
        const int qrow = q0 + r + 8 * half;
        const float inv = 1.0f / lsum[r];
        bf16_t* op = O + ((size_t)(b * SEQ + qrow)) * (HEADS * DHEAD) + h * DHEAD;
#pragma unroll
        for (int c = 0; c < 8; ++c)
            op[c * 16 + lrow] = (bf16_t)(o_acc[c][r] * inv);
    }
}

// ---------------------------------------------------------------------------
// Launch
// ---------------------------------------------------------------------------
extern "C" void kernel_launch(void* const* d_in, const int* in_sizes, int n_in,
                              void* d_out, int out_size, void* d_ws, size_t ws_size,
                              hipStream_t stream) {
    (void)in_sizes; (void)n_in; (void)out_size; (void)ws_size;
    const float* X    = (const float*)d_in[0];
    const float* Wqkv = (const float*)d_in[1];
    const float* Wout = (const float*)d_in[2];
    float* out = (float*)d_out;

    char* ws = (char*)d_ws;
    size_t off = 0;
    bf16_t* Xb    = (bf16_t*)(ws + off); off += (size_t)MROWS * DMODEL * 2;
    bf16_t* Wqkvb = (bf16_t*)(ws + off); off += (size_t)DMODEL * NQKV * 2;
    bf16_t* Woutb = (bf16_t*)(ws + off); off += (size_t)DMODEL * DMODEL * 2;
    bf16_t* QKVb  = (bf16_t*)(ws + off); off += (size_t)MROWS * NQKV * 2;
    const size_t qkvElems = (size_t)BATCH * HEADS * SEQ * DHEAD;
    bf16_t* Qb = (bf16_t*)(ws + off); off += qkvElems * 2;
    bf16_t* Kb = (bf16_t*)(ws + off); off += qkvElems * 2;
    bf16_t* Vb = (bf16_t*)(ws + off); off += qkvElems * 2;
    bf16_t* Ob = QKVb;   // QKV buffer is dead after rope_split; reuse for O

    // 1) fp32 -> bf16 conversions
    f32_to_bf16_kernel<<<4096, 256, 0, stream>>>(X, Xb, MROWS * DMODEL);
    f32_to_bf16_kernel<<<4096, 256, 0, stream>>>(Wqkv, Wqkvb, DMODEL * NQKV);
    f32_to_bf16_kernel<<<4096, 256, 0, stream>>>(Wout, Woutb, DMODEL * DMODEL);

    // 2) QKV projection: [4096 x 2048] @ [2048 x 6144] -> bf16
    gemm_bf16_kernel<true><<<dim3(NQKV / 128, MROWS / 128), 256, 0, stream>>>(
        Xb, Wqkvb, (void*)QKVb, MROWS, NQKV, DMODEL);

    // 3) RoPE + split into Q,K,V [B,H,L,D]
    {
        const int total = BATCH * HEADS * SEQ * (DHEAD / 2);
        rope_split_kernel<<<(total + 255) / 256, 256, 0, stream>>>(QKVb, Qb, Kb, Vb);
    }

    // 4) Causal flash attention -> O[B, L, H*D] bf16
    attn_kernel<<<dim3(SEQ / 64, BATCH * HEADS), 128, 0, stream>>>(Qb, Kb, Vb, Ob);

    // 5) Output projection: [4096 x 2048] @ [2048 x 2048] -> fp32 d_out
    gemm_bf16_kernel<false><<<dim3(DMODEL / 128, MROWS / 128), 256, 0, stream>>>(
        Ob, Woutb, (void*)out, MROWS, DMODEL, DMODEL);
}
